// CSDC_8246337208509
// MI455X (gfx1250) — compile-verified, matrix operand order checked by reading, not run on hardware
//
#include <hip/hip_runtime.h>
#include <cstdint>
#include <cstddef>

typedef __bf16 bf16;
typedef __attribute__((ext_vector_type(16))) __bf16 v16bf;
typedef __attribute__((ext_vector_type(8)))  float  v8f;

#define HW    65536      // 256*256
#define BN    4
#define NHEAD 8
#define HIDC  170

// ---------------------------------------------------------------------------
// LayerNorm over 64 channels (channel-first), f32 -> bf16
// ---------------------------------------------------------------------------
__global__ __launch_bounds__(256) void k_ln(const float* __restrict__ x,
                                            const float* __restrict__ w,
                                            const float* __restrict__ bb,
                                            bf16* __restrict__ out) {
  int idx = blockIdx.x * 256 + threadIdx.x;      // B*HW threads
  int b = idx >> 16, p = idx & 65535;
  const float* xp = x + ((size_t)b << 22) + p;   // b*64*HW
  float s = 0.f, s2 = 0.f;
#pragma unroll
  for (int c = 0; c < 64; ++c) {
    float v = xp[(size_t)c << 16];
    s += v; s2 = fmaf(v, v, s2);
  }
  float mu  = s * 0.015625f;
  float var = s2 * 0.015625f - mu * mu;
  float inv = rsqrtf(var + 1e-6f);
  bf16* op = out + ((size_t)b << 22) + p;
#pragma unroll
  for (int c = 0; c < 64; ++c) {
    float v = xp[(size_t)c << 16];
    op[(size_t)c << 16] = (bf16)fmaf(w[c], (v - mu) * inv, bb[c]);
  }
}

// ---------------------------------------------------------------------------
// Weight convert f32 -> bf16 with zero padding to (dstM, dstK)
// ---------------------------------------------------------------------------
__global__ __launch_bounds__(256) void k_cvt_wpad(const float* __restrict__ src,
                                                  bf16* __restrict__ dst,
                                                  int srcM, int srcK, int dstM, int dstK) {
  int idx = blockIdx.x * 256 + threadIdx.x;
  if (idx >= dstM * dstK) return;
  int m = idx / dstK, k = idx % dstK;
  dst[idx] = (m < srcM && k < srcK) ? (bf16)src[m * srcK + k] : (bf16)0.0f;
}

// ---------------------------------------------------------------------------
// WMMA bf16 GEMM: out[b][m][p] = sum_k W[(b?)m][k] * act[b][k][p]  (+res)
// Block: 256 thr = 8 waves; block tile M=32, P=128.
// Each wave: two 16x16 C tiles (m, m+16) sharing one B fragment.
// Act tile staged LDS via GLOBAL_LOAD_ASYNC_TO_LDS_B128 (ASYNCcnt path).
// ---------------------------------------------------------------------------
__global__ __launch_bounds__(256) void k_gemm(const bf16* __restrict__ W, int wStride,
                                              const bf16* __restrict__ act, int K,
                                              float* __restrict__ out, int M,
                                              const float* __restrict__ res) {
  const int b    = blockIdx.z;
  const int m0   = blockIdx.y << 5;
  const int p0   = blockIdx.x << 7;
  const int tid  = threadIdx.x;
  const int lane = tid & 31;
  const int wave = tid >> 5;
  const int pcol = (wave << 4) + (lane & 15);

  __shared__ __attribute__((aligned(16))) bf16 sB[32 * 128];  // 8 KiB act tile

  const bf16* actB  = act + (size_t)b * K * HW;
  const bf16* wrow0 = W + (size_t)(wStride ? b * wStride : 0)
                        + (size_t)(m0 + (lane & 15)) * K;
  const bf16* wrow1 = wrow0 + (size_t)16 * K;
  const int kbase = (lane & 16) ? 8 : 0;

  // LDS byte offset of this thread's two 16B chunks (flat shared ptr low 32b)
  const uint32_t ldsBase = (uint32_t)(uintptr_t)sB;

  v8f acc0 = {}, acc1 = {};
  for (int k0 = 0; k0 < K; k0 += 32) {
    __syncthreads();
    // async copy of 32x128 bf16 tile: 512 x b128, 2 per thread
#pragma unroll
    for (int i = 0; i < 2; ++i) {
      int idx   = tid + (i << 8);       // 0..511 (b128 chunk index)
      int row   = idx >> 4;             // k row 0..31
      int chunk = idx & 15;             // 16B chunk within 256B row
      uint64_t ga = (uint64_t)(uintptr_t)(actB + (size_t)(k0 + row) * HW + p0)
                    + ((uint64_t)chunk << 4);
      uint32_t la = ldsBase + ((uint32_t)idx << 4);
      asm volatile("global_load_async_to_lds_b128 %0, %1, off"
                   :: "v"(la), "v"(ga) : "memory");
    }
    if (k0 + 32 < K)   // hint next k-tile toward L2 (global_prefetch_b8)
      __builtin_prefetch(actB + (size_t)(k0 + 32 + (tid >> 3)) * HW + p0 + ((tid & 7) << 4), 0, 1);
    asm volatile("s_wait_asynccnt 0x0" ::: "memory");
    __syncthreads();

    union { v16bf v; uint32_t u[8]; } Af0, Af1;
    union { v16bf v; bf16 h[16];    } Bf;
#pragma unroll
    for (int vi = 0; vi < 8; ++vi) {
      const int kk = (vi < 4) ? (kbase + (vi << 1)) : (16 + kbase + ((vi - 4) << 1));
      Af0.u[vi]      = *reinterpret_cast<const uint32_t*>(wrow0 + k0 + kk);
      Af1.u[vi]      = *reinterpret_cast<const uint32_t*>(wrow1 + k0 + kk);
      Bf.h[2 * vi]     = sB[kk * 128 + pcol];
      Bf.h[2 * vi + 1] = sB[(kk + 1) * 128 + pcol];
    }
    acc0 = __builtin_amdgcn_wmma_f32_16x16x32_bf16(
        false, Af0.v, false, Bf.v, (short)0, acc0, false, false);
    acc1 = __builtin_amdgcn_wmma_f32_16x16x32_bf16(
        false, Af1.v, false, Bf.v, (short)0, acc1, false, false);
  }

  union { v8f v; float f[8]; } C0, C1;
  C0.v = acc0; C1.v = acc1;
  const int mo = (lane & 16) ? 8 : 0;
#pragma unroll
  for (int vi = 0; vi < 8; ++vi) {
    int m = m0 + mo + vi;
    size_t o = ((size_t)b * M + m) * HW + p0 + pcol;
    float v0 = C0.f[vi];
    size_t o1 = o + (size_t)16 * HW;
    float v1 = C1.f[vi];
    if (res) { v0 += res[o]; v1 += res[o1]; }
    out[o]  = v0;
    out[o1] = v1;
  }
}

// ---------------------------------------------------------------------------
// Depthwise 3x3 SAME (zero pad). mode 1: out = tanh(conv) + in
// ---------------------------------------------------------------------------
__global__ __launch_bounds__(256) void k_dwconv(const float* __restrict__ in,
                                                const float* __restrict__ w9,
                                                float* __restrict__ out,
                                                int C, int CstrIn, int chOffIn,
                                                int CstrOut, int chOffOut, int mode) {
  int idx = blockIdx.x * 256 + threadIdx.x;  // B*C*HW threads (exact)
  int p = idx & 65535;
  int bc = idx >> 16;
  int c = bc % C, b = bc / C;
  int xx0 = p & 255, yy0 = p >> 8;
  const float* ip = in + ((size_t)b * CstrIn + chOffIn + c) * HW;
  const float* wc = w9 + (size_t)c * 9;
  float s = 0.f;
#pragma unroll
  for (int dy = -1; dy <= 1; ++dy) {
#pragma unroll
    for (int dx = -1; dx <= 1; ++dx) {
      int yy = yy0 + dy, xx = xx0 + dx;
      if (yy >= 0 && yy < 256 && xx >= 0 && xx < 256)
        s = fmaf(wc[(dy + 1) * 3 + dx + 1], ip[yy * 256 + xx], s);
    }
  }
  float o = (mode == 1) ? (tanhf(s) + ip[p]) : s;
  out[((size_t)b * CstrOut + chOffOut + c) * HW + p] = o;
}

// ---------------------------------------------------------------------------
// Per-channel-plane L2 norm (l2norm over hw axis); row = b*64 + c
// ---------------------------------------------------------------------------
__global__ __launch_bounds__(256) void k_rownorm(const float* __restrict__ src,
                                                 int Ctot, float* __restrict__ norms) {
  int row = blockIdx.x;
  int b = row >> 6, c = row & 63;
  const float* p = src + ((size_t)b * Ctot + c) * HW;
  float s = 0.f;
  for (int i = threadIdx.x; i < HW; i += 256) { float v = p[i]; s = fmaf(v, v, s); }
  __shared__ float r[256];
  r[threadIdx.x] = s; __syncthreads();
  for (int st = 128; st > 0; st >>= 1) {
    if (threadIdx.x < st) r[threadIdx.x] += r[threadIdx.x + st];
    __syncthreads();
  }
  if (threadIdx.x == 0) norms[row] = fmaxf(sqrtf(r[0]), 1e-12f);
}

// ---------------------------------------------------------------------------
// Attention logits + softmax: one block per (b,h); 8x8 logits over hw=65536.
// l2norm folded in as post-scale 1/(|q||k|); temperature applied pre-softmax.
// ---------------------------------------------------------------------------
__global__ __launch_bounds__(256) void k_attn(const float* __restrict__ q, int qC,
                                              const float* __restrict__ k, int kC,
                                              const float* __restrict__ qn,
                                              const float* __restrict__ kn,
                                              const float* __restrict__ temp,
                                              float* __restrict__ attn) {
  const int bh = blockIdx.x;
  const int b = bh >> 3, h = bh & 7;
  const int tid = threadIdx.x;
  const float* qb = q + ((size_t)b * qC + h * 8) * HW;
  const float* kb = k + ((size_t)b * kC + h * 8) * HW;

  float acc[8][8] = {};
  for (int e = tid; e < HW; e += 256) {
    float qv[8], kv[8];
#pragma unroll
    for (int i = 0; i < 8; ++i) { qv[i] = qb[(size_t)i * HW + e]; kv[i] = kb[(size_t)i * HW + e]; }
#pragma unroll
    for (int i = 0; i < 8; ++i)
#pragma unroll
      for (int j = 0; j < 8; ++j) acc[i][j] = fmaf(qv[i], kv[j], acc[i][j]);
  }

  __shared__ float red[32 * 256];
  __shared__ float logits[64];
#pragma unroll
  for (int half = 0; half < 2; ++half) {
#pragma unroll
    for (int e = 0; e < 32; ++e) {
      int ent = half * 32 + e;
      red[e * 256 + tid] = acc[ent >> 3][ent & 7];
    }
    __syncthreads();
    for (int s = 128; s > 0; s >>= 1) {
      for (int e2 = tid; e2 < 32 * s; e2 += 256) {
        int ent = e2 / s, pos = e2 % s;
        red[ent * 256 + pos] += red[ent * 256 + pos + s];
      }
      __syncthreads();
    }
    if (tid < 32) logits[half * 32 + tid] = red[tid * 256];
    __syncthreads();
  }
  if (tid < 64) {
    int i = tid >> 3, j = tid & 7;
    logits[tid] = logits[tid] * temp[h] / (qn[b * 64 + h * 8 + i] * kn[b * 64 + h * 8 + j]);
  }
  __syncthreads();
  if (tid < 8) {
    int i = tid;
    float mx = -1e30f;
#pragma unroll
    for (int j = 0; j < 8; ++j) mx = fmaxf(mx, logits[i * 8 + j]);
    float e[8], s = 0.f;
#pragma unroll
    for (int j = 0; j < 8; ++j) { e[j] = __expf(logits[i * 8 + j] - mx); s += e[j]; }
    float inv = 1.0f / s;
#pragma unroll
    for (int j = 0; j < 8; ++j) attn[(size_t)bh * 64 + i * 8 + j] = e[j] * inv;
  }
}

// ---------------------------------------------------------------------------
// Fold attn into w_proj:  Mf[b][o][h*8+d] = sum_cc wproj[o][h*8+cc]*attn[b][h][cc][d]
// ---------------------------------------------------------------------------
__global__ __launch_bounds__(256) void k_fuse_proj(const float* __restrict__ wproj,
                                                   const float* __restrict__ attn,
                                                   bf16* __restrict__ Mf) {
  int idx = blockIdx.x * 256 + threadIdx.x;  // B*64*64
  int b = idx >> 12, r = idx & 4095;
  int o = r >> 6, hd = r & 63, h = hd >> 3, d = hd & 7;
  float s = 0.f;
#pragma unroll
  for (int cc = 0; cc < 8; ++cc)
    s = fmaf(wproj[o * 64 + h * 8 + cc], attn[(((size_t)(b * 8 + h) * 8 + cc) * 8) + d], s);
  Mf[idx] = (bf16)s;
}

// ---------------------------------------------------------------------------
// Extract v (= kvdw channels 64..127) -> bf16 64-ch tensor
// ---------------------------------------------------------------------------
__global__ __launch_bounds__(256) void k_extract_v(const float* __restrict__ kvdw,
                                                   bf16* __restrict__ dst) {
  int idx = blockIdx.x * 256 + threadIdx.x;  // B*64*HW
  int p = idx & 65535;
  int bc = idx >> 16;
  int c = bc & 63, b = bc >> 6;
  dst[idx] = (bf16)kvdw[((size_t)b * 128 + 64 + c) * HW + p];
}

// ---------------------------------------------------------------------------
// x1*x2 -> bf16 with zero pad to 192 channels (for K=192 WMMA GEMM)
// ---------------------------------------------------------------------------
__global__ __launch_bounds__(256) void k_prod_pad(const float* __restrict__ t,
                                                  bf16* __restrict__ dst) {
  int idx = blockIdx.x * 256 + threadIdx.x;  // B*192*HW
  int p = idx & 65535;
  int bc = idx >> 16;
  int c = bc % 192, b = bc / 192;
  bf16 o = (bf16)0.0f;
  if (c < HIDC) {
    float a  = t[((size_t)b * 340 + c) * HW + p];
    float b2 = t[((size_t)b * 340 + HIDC + c) * HW + p];
    o = (bf16)(a * b2);
  }
  dst[idx] = o;
}

// ---------------------------------------------------------------------------
extern "C" void kernel_launch(void* const* d_in, const int* in_sizes, int n_in,
                              void* d_out, int out_size, void* d_ws, size_t ws_size,
                              hipStream_t stream) {
  const float* x      = (const float*)d_in[0];
  const float* y      = (const float*)d_in[1];
  const float* lnw    = (const float*)d_in[2];
  const float* lnb    = (const float*)d_in[3];
  const float* temp   = (const float*)d_in[4];
  const float* wq     = (const float*)d_in[5];
  const float* wq_dw  = (const float*)d_in[6];
  const float* wkv    = (const float*)d_in[7];
  const float* wkv_dw = (const float*)d_in[8];
  const float* wproj  = (const float*)d_in[9];
  const float* w_in   = (const float*)d_in[10];
  const float* w_dw   = (const float*)d_in[11];
  const float* w_dw1  = (const float*)d_in[12];
  const float* w_dw2  = (const float*)d_in[13];
  const float* w_out  = (const float*)d_in[14];
  float* out = (float*)d_out;

  char* ws = (char*)d_ws;
  size_t off = 0;
  auto alloc = [&](size_t bytes) -> char* {
    char* p = ws + off;
    off = (off + bytes + 255) & ~(size_t)255;
    return p;
  };
  // bf16 activation region: 3 consecutive 64-ch slots == one 192-ch slot (reused)
  bf16* A0 = (bf16*)alloc((size_t)64 * BN * HW * 2);   // lnx -> ln2 -> x1x2[0:64)
  bf16* A1 = (bf16*)alloc((size_t)64 * BN * HW * 2);   // lny
  bf16* A2 = (bf16*)alloc((size_t)64 * BN * HW * 2);   // v_bf16
  // f32 region (F0..F3 contiguous 384 MiB -> reused as 352-ch h and 340-ch t)
  float* F0 = (float*)alloc((size_t)64 * BN * HW * 4);   // q conv
  float* F1 = (float*)alloc((size_t)128 * BN * HW * 4);  // kv conv
  float* F2 = (float*)alloc((size_t)64 * BN * HW * 4);   // q dwconv
  float* F3 = (float*)alloc((size_t)128 * BN * HW * 4);  // kv dwconv
  float* F4 = (float*)alloc((size_t)64 * BN * HW * 4);   // xcab (residual base)
  float* G1 = (float*)alloc((size_t)340 * BN * HW * 4);  // d = dwconv(h)
  float* qn   = (float*)alloc(256 * 4);
  float* kn   = (float*)alloc(256 * 4);
  float* attn = (float*)alloc((size_t)BN * NHEAD * 64 * 4);
  bf16* Mf     = (bf16*)alloc((size_t)BN * 64 * 64 * 2);
  bf16* wq_b   = (bf16*)alloc(64 * 64 * 2);
  bf16* wkv_b  = (bf16*)alloc(128 * 64 * 2);
  bf16* win_b  = (bf16*)alloc(352 * 64 * 2);
  bf16* wout_b = (bf16*)alloc(64 * 192 * 2);
  float* H352 = F0;  // 352-ch f32, overlays F0..F3
  float* T340 = F0;  // 340-ch f32, overlays F0..F3 (h dead by then)

  // --- weight conversion (bf16, zero-padded to WMMA-tileable shapes) ---
  k_cvt_wpad<<<(64 * 64 + 255) / 256, 256, 0, stream>>>(wq, wq_b, 64, 64, 64, 64);
  k_cvt_wpad<<<(128 * 64 + 255) / 256, 256, 0, stream>>>(wkv, wkv_b, 128, 64, 128, 64);
  k_cvt_wpad<<<(352 * 64 + 255) / 256, 256, 0, stream>>>(w_in, win_b, 340, 64, 352, 64);
  k_cvt_wpad<<<(64 * 192 + 255) / 256, 256, 0, stream>>>(w_out, wout_b, 64, 170, 64, 192);

  // --- CAB ---
  k_ln<<<1024, 256, 0, stream>>>(x, lnw, lnb, A0);
  k_ln<<<1024, 256, 0, stream>>>(y, lnw, lnb, A1);
  k_gemm<<<dim3(512, 2, BN), 256, 0, stream>>>(wq_b, 0, A0, 64, F0, 64, nullptr);
  k_gemm<<<dim3(512, 4, BN), 256, 0, stream>>>(wkv_b, 0, A1, 64, F1, 128, nullptr);
  k_dwconv<<<(BN * 64 * HW) / 256, 256, 0, stream>>>(F0, wq_dw, F2, 64, 64, 0, 64, 0, 0);
  k_dwconv<<<(BN * 128 * HW) / 256, 256, 0, stream>>>(F1, wkv_dw, F3, 128, 128, 0, 128, 0, 0);
  k_rownorm<<<256, 256, 0, stream>>>(F2, 64, qn);
  k_rownorm<<<256, 256, 0, stream>>>(F3, 128, kn);
  k_attn<<<BN * NHEAD, 256, 0, stream>>>(F2, 64, F3, 128, qn, kn, temp, attn);
  k_extract_v<<<(BN * 64 * HW) / 256, 256, 0, stream>>>(F3, A2);
  k_fuse_proj<<<(BN * 64 * 64) / 256, 256, 0, stream>>>(wproj, attn, Mf);
  // xcab = x + (Mf_b . v_b)   -- attention+projection as one per-batch WMMA GEMM
  k_gemm<<<dim3(512, 2, BN), 256, 0, stream>>>(Mf, 64 * 64, A2, 64, F4, 64, x);

  // --- IEL ---
  k_ln<<<1024, 256, 0, stream>>>(F4, lnw, lnb, A0);
  k_gemm<<<dim3(512, 11, BN), 256, 0, stream>>>(win_b, 0, A0, 64, H352, 352, nullptr);
  k_dwconv<<<(BN * 340 * HW) / 256, 256, 0, stream>>>(H352, w_dw, G1, 340, 352, 0, 340, 0, 0);
  k_dwconv<<<(BN * 170 * HW) / 256, 256, 0, stream>>>(G1, w_dw1, T340, 170, 340, 0, 340, 0, 1);
  k_dwconv<<<(BN * 170 * HW) / 256, 256, 0, stream>>>(G1, w_dw2, T340, 170, 340, 170, 340, 170, 1);
  k_prod_pad<<<(BN * 192 * HW) / 256, 256, 0, stream>>>(T340, A0);  // 192-ch bf16 overlay
  // out = xcab + W_out . (x1*x2)
  k_gemm<<<dim3(512, 2, BN), 256, 0, stream>>>(wout_b, 0, (const bf16*)A0, 192, out, 64, F4);
  (void)in_sizes; (void)n_in; (void)out_size; (void)ws_size; (void)A1; (void)A2;
}
